// JambaMambaDecoderLayer_10445360464009
// MI455X (gfx1250) — matrix operand order repaired and checked by reading, not compiled, not run on hardware
//
#include <hip/hip_runtime.h>
#include <hip/hip_bf16.h>
#include <math.h>

// ---------------------------------------------------------------------------
// Jamba/Mamba decoder layer for MI455X (gfx1250, wave32, WMMA).
// GEMMs are compute-bound (AI ~520 FLOP/B vs machine balance ~215 FLOP/B at
// 23.3 TB/s), so they run on v_wmma_f32_16x16x32_bf16 with on-the-fly
// fp32->bf16 conversion. Global->LDS staging is software-pipelined through
// registers so the s_wait_loadcnt for tile t+1 lands after the WMMAs of tile t.
// ---------------------------------------------------------------------------

typedef __bf16 bf16_t;
typedef __bf16 v16bf __attribute__((ext_vector_type(16)));
typedef __bf16 v8bf  __attribute__((ext_vector_type(8)));
typedef float  v8f   __attribute__((ext_vector_type(8)));

#define L_SEQ 1024
#define H_DIM 2048
#define DI    4096
#define DT_R  128
#define N_ST  16
#define KCONV 4
#define F_FF  8192
#define EPSV  1e-6f

__device__ __forceinline__ float siluf(float x)     { return x / (1.f + __expf(-x)); }
__device__ __forceinline__ float softplusf(float x) { return (x > 20.f) ? x : log1pf(__expf(x)); }

// ---------------------------------------------------------------------------
// Generic WMMA GEMM:  C[M,N] = A[M,K] @ B[N,K]^T   (A,B row-major, K contig)
// MODE 0: C = acc
// MODE 1: C = softplus(acc + bias[n])          (dt_proj)
// MODE 2: C = acc + res[m*N + n]               (residual adds)
// Block tile 64(M) x 128(N) x 32(K); 256 threads = 8 waves (2x4),
// each wave owns a 32x32 patch = 2x2 v_wmma_f32_16x16x32_bf16 tiles.
// ---------------------------------------------------------------------------
namespace {

constexpr int BM = 64, BN = 128, BK = 32, LDSS = 40; // 40-bf16 stride: conflict-free b128

__device__ __forceinline__ void load_tile_regs(
    const float* __restrict__ A, const float* __restrict__ B,
    int N, int K, int m0, int n0, int k0, int lrow, int lcol,
    float4 (&ra)[2], float4 (&rb)[4])
{
    // Issue all loads before any use so they pipeline (no early waits).
    #pragma unroll
    for (int i = 0; i < 2; i++)
        ra[i] = *(const float4*)(A + (size_t)(m0 + lrow + i * 32) * K + k0 + lcol);
    if (n0 + BN <= N) {                      // uniform fast path (all but ragged x_proj tile)
        #pragma unroll
        for (int i = 0; i < 4; i++)
            rb[i] = *(const float4*)(B + (size_t)(n0 + lrow + i * 32) * K + k0 + lcol);
    } else {
        #pragma unroll
        for (int i = 0; i < 4; i++) {
            const int r = lrow + i * 32;
            rb[i] = make_float4(0.f, 0.f, 0.f, 0.f);
            if (n0 + r < N)
                rb[i] = *(const float4*)(B + (size_t)(n0 + r) * K + k0 + lcol);
        }
    }
}

__device__ __forceinline__ void store_tile_lds(
    bf16_t* __restrict__ sA, bf16_t* __restrict__ sB, int lrow, int lcol,
    const float4 (&ra)[2], const float4 (&rb)[4])
{
    #pragma unroll
    for (int i = 0; i < 2; i++) {
        bf16_t* d = &sA[(lrow + i * 32) * LDSS + lcol];
        d[0] = (bf16_t)ra[i].x; d[1] = (bf16_t)ra[i].y;
        d[2] = (bf16_t)ra[i].z; d[3] = (bf16_t)ra[i].w;
    }
    #pragma unroll
    for (int i = 0; i < 4; i++) {
        bf16_t* d = &sB[(lrow + i * 32) * LDSS + lcol];
        d[0] = (bf16_t)rb[i].x; d[1] = (bf16_t)rb[i].y;
        d[2] = (bf16_t)rb[i].z; d[3] = (bf16_t)rb[i].w;
    }
}

} // namespace

template<int MODE>
__global__ void __launch_bounds__(256)
gemm_wmma_bf16(const float* __restrict__ A, const float* __restrict__ B,
               float* __restrict__ C, const float* __restrict__ extra,
               int M, int N, int K)
{
    __shared__ bf16_t sA[BM * LDSS];
    __shared__ bf16_t sB[BN * LDSS];

    const int tid  = threadIdx.x;
    const int lane = tid & 31;
    const int wave = tid >> 5;
    const int wm   = wave >> 2;          // 0..1  (M direction)
    const int wn   = wave & 3;           // 0..3  (N direction)
    const int m0   = blockIdx.y * BM;
    const int n0   = blockIdx.x * BN;

    v8f acc[2][2];
    #pragma unroll
    for (int i = 0; i < 2; i++)
        #pragma unroll
        for (int j = 0; j < 2; j++)
            acc[i][j] = (v8f){0.f,0.f,0.f,0.f,0.f,0.f,0.f,0.f};

    const int lrow = tid >> 3;           // 0..31 (tile row for staging)
    const int lcol = (tid & 7) * 4;      // 0,4,...,28

    const int hh  = lane >> 4;           // lane half: 0 or 1
    const int lr  = lane & 15;
    const int kbA = hh * 8;              // A frag per-lane K base (ISA 16-bit A layout)
    const int kbB = hh * 8;              // B frag per-lane K base (same K order as A)

    // ---- software pipeline: prologue load of tile 0 ----
    float4 ra[2], rb[4], ra2[2], rb2[4];
    load_tile_regs(A, B, N, K, m0, n0, 0, lrow, lcol, ra, rb);

    for (int k0 = 0; k0 < K; k0 += BK) {
        // publish tile t to LDS (this is where the loadcnt wait lands)
        store_tile_lds(sA, sB, lrow, lcol, ra, rb);
        __syncthreads();

        // kick off global loads of tile t+1; no consumer until next iteration
        const bool more = (k0 + BK < K);
        if (more)
            load_tile_regs(A, B, N, K, m0, n0, k0 + BK, lrow, lcol, ra2, rb2);

        // fragments + 4 WMMAs for tile t (overlaps with tile t+1 loads)
        union FragU { v16bf v; v8bf h[2]; };
        FragU a[2], b[2];
        #pragma unroll
        for (int t = 0; t < 2; t++) {
            const bf16_t* p = &sA[(wm * 32 + t * 16 + lr) * LDSS];
            a[t].h[0] = *(const v8bf*)(p + kbA);        // K = kb..kb+7
            a[t].h[1] = *(const v8bf*)(p + 16 + kbA);   // K = 16+kb..16+kb+7
        }
        #pragma unroll
        for (int t = 0; t < 2; t++) {
            const bf16_t* p = &sB[(wn * 32 + t * 16 + lr) * LDSS];
            b[t].h[0] = *(const v8bf*)(p + kbB);        // K = kb..kb+7
            b[t].h[1] = *(const v8bf*)(p + 16 + kbB);   // K = 16+kb..16+kb+7
        }
        #pragma unroll
        for (int i = 0; i < 2; i++)
            #pragma unroll
            for (int j = 0; j < 2; j++)
                acc[i][j] = __builtin_amdgcn_wmma_f32_16x16x32_bf16(
                    false, a[i].v, false, b[j].v, (short)0, acc[i][j], false, false);
        __syncthreads();

        if (more) {
            #pragma unroll
            for (int i = 0; i < 2; i++) ra[i] = ra2[i];
            #pragma unroll
            for (int i = 0; i < 4; i++) rb[i] = rb2[i];
        }
    }

    // ---- epilogue (C/D layout: m = r + 8*(lane/16), n = lane%16) ----
    const int mlane = hh * 8;
    #pragma unroll
    for (int i = 0; i < 2; i++) {
        #pragma unroll
        for (int j = 0; j < 2; j++) {
            const int ng = n0 + wn * 32 + j * 16 + lr;
            if (ng >= N) continue;
            #pragma unroll
            for (int r = 0; r < 8; r++) {
                const int mg = m0 + wm * 32 + i * 16 + mlane + r;
                float v = acc[i][j][r];
                if (MODE == 1) v = softplusf(v + extra[ng]);
                if (MODE == 2) v = v + extra[(size_t)mg * N + ng];
                C[(size_t)mg * N + ng] = v;
            }
        }
    }
}

// ---------------------------------------------------------------------------
// RMSNorm over rows of length D (block per row, 8 waves)
// ---------------------------------------------------------------------------
__global__ void __launch_bounds__(256)
rmsnorm_kernel(const float* __restrict__ x, const float* __restrict__ w,
               float* __restrict__ out, int D)
{
    const int row = blockIdx.x;
    const float* xr = x + (size_t)row * D;
    float ss = 0.f;
    for (int i = threadIdx.x; i < D; i += 256) { const float v = xr[i]; ss += v * v; }
    #pragma unroll
    for (int off = 16; off > 0; off >>= 1) ss += __shfl_xor(ss, off, 32);
    __shared__ float s_part[8];
    if ((threadIdx.x & 31) == 0) s_part[threadIdx.x >> 5] = ss;
    __syncthreads();
    float tot = 0.f;
    #pragma unroll
    for (int i = 0; i < 8; i++) tot += s_part[i];
    const float scale = rsqrtf(tot / (float)D + EPSV);
    for (int i = threadIdx.x; i < D; i += 256)
        out[(size_t)row * D + i] = xr[i] * scale * w[i];
}

// ---------------------------------------------------------------------------
// Causal depthwise conv (K=4) + bias + SiLU.  proj row stride = 2*DI, h = cols [0,DI)
// ---------------------------------------------------------------------------
__global__ void __launch_bounds__(256)
conv_silu_kernel(const float* __restrict__ proj, const float* __restrict__ cw,
                 const float* __restrict__ cb, float* __restrict__ h)
{
    const int idx = blockIdx.x * 256 + threadIdx.x;
    if (idx >= L_SEQ * DI) return;
    const int l = idx / DI, d = idx % DI;
    float acc = cb[d];
    #pragma unroll
    for (int k = 0; k < KCONV; k++) {
        const int li = l - (KCONV - 1) + k;
        if (li >= 0) acc += proj[(size_t)li * (2 * DI) + d] * cw[d * KCONV + k];
    }
    h[idx] = siluf(acc);
}

// ---------------------------------------------------------------------------
// RMSNorm of the three ssm slices (dt:128, B:16, C:16). Block=160 (5 waves) per row.
// ---------------------------------------------------------------------------
__global__ void ssm_norm_kernel(const float* __restrict__ ssm,
                                const float* __restrict__ dtw,
                                const float* __restrict__ bw,
                                const float* __restrict__ cw,
                                float* __restrict__ dtn,
                                float* __restrict__ Bn,
                                float* __restrict__ Cn)
{
    const int row = blockIdx.x;
    const int t = threadIdx.x;                 // 0..159
    __shared__ float s_acc[3];
    if (t < 3) s_acc[t] = 0.f;
    __syncthreads();
    const float v = ssm[row * 160 + t];
    const int seg = (t < 128) ? 0 : (t < 144 ? 1 : 2);
    atomicAdd(&s_acc[seg], v * v);
    __syncthreads();
    if (t < 128)
        dtn[row * 128 + t] = v * rsqrtf(s_acc[0] / 128.f + EPSV) * dtw[t];
    else if (t < 144)
        Bn[row * 16 + (t - 128)] = v * rsqrtf(s_acc[1] / 16.f + EPSV) * bw[t - 128];
    else
        Cn[row * 16 + (t - 144)] = v * rsqrtf(s_acc[2] / 16.f + EPSV) * cw[t - 144];
}

// ---------------------------------------------------------------------------
// Selective scan. One thread per (d,n) state; 16-lane shuffle reduction over N.
// y[l,d] = (sum_n s[l,d,n]*C[l,n] + h[l,d]*D[d]) * silu(gate[l,d])
// ---------------------------------------------------------------------------
__global__ void __launch_bounds__(256)
scan_kernel(const float* __restrict__ delta, const float* __restrict__ hbuf,
            const float* __restrict__ Bn, const float* __restrict__ Cn,
            const float* __restrict__ Amat, const float* __restrict__ Dvec,
            const float* __restrict__ proj /* gate at col offset DI */,
            float* __restrict__ y)
{
    const int g = blockIdx.x * 256 + threadIdx.x;
    const int d = g >> 4;
    const int n = g & 15;
    if (d >= DI) return;
    const float a  = Amat[d * N_ST + n];
    const float Dd = Dvec[d];
    float s = 0.f;
    for (int l = 0; l < L_SEQ; l++) {
        const float dl = delta[(size_t)l * DI + d];
        const float hl = hbuf[(size_t)l * DI + d];
        const float bl = Bn[l * N_ST + n];
        const float cl = Cn[l * N_ST + n];
        s = __expf(dl * a) * s + dl * bl * hl;
        float p = s * cl;
        #pragma unroll
        for (int off = 1; off < 16; off <<= 1) p += __shfl_xor(p, off, 32);
        if (n == 0) {
            const float gate = proj[(size_t)l * (2 * DI) + DI + d];
            y[(size_t)l * DI + d] = (p + hl * Dd) * siluf(gate);
        }
    }
}

// ---------------------------------------------------------------------------
// SwiGLU: out[l,f] = silu(gu[l,f]) * gu[l,F+f]
// ---------------------------------------------------------------------------
__global__ void __launch_bounds__(256)
swiglu_kernel(const float* __restrict__ gu, float* __restrict__ out)
{
    const int idx = blockIdx.x * 256 + threadIdx.x;
    if (idx >= L_SEQ * F_FF) return;
    const int l = idx / F_FF, f = idx % F_FF;
    const float g = gu[(size_t)l * (2 * F_FF) + f];
    const float u = gu[(size_t)l * (2 * F_FF) + F_FF + f];
    out[idx] = siluf(g) * u;
}

// ---------------------------------------------------------------------------
extern "C" void kernel_launch(void* const* d_in, const int* in_sizes, int n_in,
                              void* d_out, int out_size, void* d_ws, size_t ws_size,
                              hipStream_t stream)
{
    (void)in_sizes; (void)n_in; (void)out_size; (void)ws_size;

    const float* hs        = (const float*)d_in[0];   // (L,H)
    const float* in_ln_w   = (const float*)d_in[1];   // (H)
    const float* ff_ln_w   = (const float*)d_in[2];   // (H)
    const float* in_proj_w = (const float*)d_in[3];   // (2*Di, H)
    const float* conv_w    = (const float*)d_in[4];   // (Di, 4)
    const float* conv_b    = (const float*)d_in[5];   // (Di)
    const float* x_proj_w  = (const float*)d_in[6];   // (160, Di)
    const float* dt_ln_w   = (const float*)d_in[7];   // (128)
    const float* b_ln_w    = (const float*)d_in[8];   // (16)
    const float* c_ln_w    = (const float*)d_in[9];   // (16)
    const float* dt_proj_w = (const float*)d_in[10];  // (Di, 128)
    const float* dt_proj_b = (const float*)d_in[11];  // (Di)
    const float* A_mat     = (const float*)d_in[12];  // (Di, 16)
    const float* D_vec     = (const float*)d_in[13];  // (Di)
    const float* out_proj_w= (const float*)d_in[14];  // (H, Di)
    const float* gate_up_w = (const float*)d_in[15];  // (2F, H)
    const float* down_w    = (const float*)d_in[16];  // (H, F)
    float* out = (float*)d_out;                       // (L,H)

    // workspace layout (fp32)
    float* ws = (float*)d_ws;
    size_t off = 0;
    auto alloc = [&](size_t n) { float* p = ws + off; off += n; return p; };
    float* xn    = alloc((size_t)L_SEQ * H_DIM);      // rmsnorm out (reused for 2nd norm)
    float* proj  = alloc((size_t)L_SEQ * 2 * DI);     // in_proj out (h_raw | gate)
    float* hbuf  = alloc((size_t)L_SEQ * DI);         // conv+silu out
    float* ssm   = alloc((size_t)L_SEQ * 160);
    float* dtn   = alloc((size_t)L_SEQ * DT_R);
    float* Bn    = alloc((size_t)L_SEQ * N_ST);
    float* Cn    = alloc((size_t)L_SEQ * N_ST);
    float* delta = alloc((size_t)L_SEQ * DI);
    float* ybuf  = alloc((size_t)L_SEQ * DI);
    float* res2  = alloc((size_t)L_SEQ * H_DIM);
    float* gu    = alloc((size_t)L_SEQ * 2 * F_FF);
    float* swb   = alloc((size_t)L_SEQ * F_FF);

    const dim3 blk(256);

    // 1) x = rmsnorm(hidden, in_ln_w)
    rmsnorm_kernel<<<L_SEQ, blk, 0, stream>>>(hs, in_ln_w, xn, H_DIM);

    // 2) proj = x @ in_proj_w.T          (1024 x 8192, K=2048)
    gemm_wmma_bf16<0><<<dim3(2 * DI / 128, L_SEQ / 64), blk, 0, stream>>>(
        xn, in_proj_w, proj, nullptr, L_SEQ, 2 * DI, H_DIM);

    // 3) h = silu(causal_conv(proj[:, :Di]) + b)
    conv_silu_kernel<<<(L_SEQ * DI + 255) / 256, blk, 0, stream>>>(proj, conv_w, conv_b, hbuf);

    // 4) ssm = h @ x_proj_w.T            (1024 x 160, K=4096)
    gemm_wmma_bf16<0><<<dim3((160 + 127) / 128, L_SEQ / 64), blk, 0, stream>>>(
        hbuf, x_proj_w, ssm, nullptr, L_SEQ, 160, DI);

    // 5) dt/B/C rmsnorms
    ssm_norm_kernel<<<L_SEQ, 160, 0, stream>>>(ssm, dt_ln_w, b_ln_w, c_ln_w, dtn, Bn, Cn);

    // 6) delta = softplus(dtn @ dt_proj_w.T + b)   (1024 x 4096, K=128)
    gemm_wmma_bf16<1><<<dim3(DI / 128, L_SEQ / 64), blk, 0, stream>>>(
        dtn, dt_proj_w, delta, dt_proj_b, L_SEQ, DI, DT_R);

    // 7) selective scan -> y (gated)
    scan_kernel<<<(DI * N_ST) / 256, blk, 0, stream>>>(
        delta, hbuf, Bn, Cn, A_mat, D_vec, proj, ybuf);

    // 8) res2 = hidden + y @ out_proj_w.T          (1024 x 2048, K=4096)
    gemm_wmma_bf16<2><<<dim3(H_DIM / 128, L_SEQ / 64), blk, 0, stream>>>(
        ybuf, out_proj_w, res2, hs, L_SEQ, H_DIM, DI);

    // 9) x2 = rmsnorm(res2, pre_ff_ln_w)  (reuse xn)
    rmsnorm_kernel<<<L_SEQ, blk, 0, stream>>>(res2, ff_ln_w, xn, H_DIM);

    // 10) gu = x2 @ gate_up_w.T           (1024 x 16384, K=2048)
    gemm_wmma_bf16<0><<<dim3(2 * F_FF / 128, L_SEQ / 64), blk, 0, stream>>>(
        xn, gate_up_w, gu, nullptr, L_SEQ, 2 * F_FF, H_DIM);

    // 11) swb = silu(gu[:, :F]) * gu[:, F:]
    swiglu_kernel<<<(L_SEQ * F_FF + 255) / 256, blk, 0, stream>>>(gu, swb);

    // 12) out = res2 + swb @ down_w.T     (1024 x 2048, K=8192)
    gemm_wmma_bf16<2><<<dim3(H_DIM / 128, L_SEQ / 64), blk, 0, stream>>>(
        swb, down_w, out, res2, L_SEQ, H_DIM, F_FF);
}
